// AttentionStream_7481833030171
// MI455X (gfx1250) — matrix operand order repaired and checked by reading, not compiled, not run on hardware
//
#include <hip/hip_runtime.h>
#include <stdint.h>

typedef __attribute__((ext_vector_type(16))) __bf16          v16bf;
typedef __attribute__((ext_vector_type(8)))  float           v8f;
typedef __attribute__((ext_vector_type(16))) unsigned short  ushort16;
typedef __attribute__((ext_vector_type(8)))  unsigned short  ushort8;

#define NB     4
#define NSEQ   2048
#define NROWS  (NB * NSEQ)      // 8192
#define DIM    512
#define INNER  512
#define HEADS  8
#define DHEAD  64
#define SCALE  0.125f           // 64^-0.5
#define NREL   1025

// ---------- helpers ----------

__device__ __forceinline__ unsigned short f2bf(float f) {
  unsigned u = __builtin_bit_cast(unsigned, f);
  u += 0x7FFFu + ((u >> 16) & 1u);     // round-to-nearest-even
  return (unsigned short)(u >> 16);
}

__device__ __forceinline__ v16bf make_frag(ushort8 lo, ushort8 hi) {
  ushort16 u;
#pragma unroll
  for (int i = 0; i < 8; ++i) { u[i] = lo[i]; u[8 + i] = hi[i]; }
  return __builtin_bit_cast(v16bf, u);
}

// A-matrix fragment (16x32 bf16): row = row0 + (lane&15).
// Layout: lanes 0-15 hold K 0-15, lanes 16-31 hold K 16-31 (contiguous).
__device__ __forceinline__ v16bf load_a(const unsigned short* __restrict__ base,
                                        int ld, int row0, int k0) {
  int lane = threadIdx.x & 31;
  const unsigned short* p =
      base + (size_t)(row0 + (lane & 15)) * ld + k0 + ((lane >> 4) << 4);
  ushort8 lo = *(const ushort8*)p;
  ushort8 hi = *(const ushort8*)(p + 8);
  return make_frag(lo, hi);
}

// B-matrix fragment (32x16 bf16) for D = A*B.  Storage: each matmul-column is a
// contiguous row of length `ld` along K.  Lane = column, lane-half selects K 0-15/16-31.
__device__ __forceinline__ v16bf load_b(const unsigned short* __restrict__ base,
                                        int ld, int col0, int k0) {
  int lane = threadIdx.x & 31;
  const unsigned short* p =
      base + (size_t)(col0 + (lane & 15)) * ld + k0 + ((lane >> 4) << 4);
  ushort16 u = *(const ushort16*)p;
  return __builtin_bit_cast(v16bf, u);
}

// B fragment from rel_emb (row index clamped to [0,1024]); rel stored [1025][64] bf16.
__device__ __forceinline__ v16bf load_rel(const unsigned short* __restrict__ relb,
                                          int D0, int w0, int k0) {
  int lane = threadIdx.x & 31;
  int idx = D0 + w0 + (lane & 15);
  idx = idx < 0 ? 0 : (idx > 1024 ? 1024 : idx);
  const unsigned short* p = relb + (size_t)idx * 64 + k0 + ((lane >> 4) << 4);
  ushort16 u = *(const ushort16*)p;
  return __builtin_bit_cast(v16bf, u);
}

__device__ __forceinline__ v8f wmma_bf16(v16bf a, v16bf b, v8f c) {
  return __builtin_amdgcn_wmma_f32_16x16x32_bf16(false, a, false, b,
                                                 (short)0, c, false, false);
}

__device__ __forceinline__ v8f zero8() {
  v8f z = {0.f, 0.f, 0.f, 0.f, 0.f, 0.f, 0.f, 0.f};
  return z;
}

// ---------- conversion kernels ----------

__global__ void k_f2bf(const float* __restrict__ s, unsigned short* __restrict__ d, int n) {
  int i = blockIdx.x * blockDim.x + threadIdx.x;
  if (i < n) d[i] = f2bf(s[i]);
}

// s[R][C] (row-major) -> d[C][R] bf16 (transpose)
__global__ void k_f2bfT(const float* __restrict__ s, unsigned short* __restrict__ d,
                        int R, int C) {
  int i = blockIdx.x * blockDim.x + threadIdx.x;
  if (i < R * C) {
    int r = i / C, c = i % C;
    d[(size_t)c * R + r] = f2bf(s[i]);
  }
}

// ---------- QKV projection: [8192,512] x [512,1536] ----------
// One wave per 16x16 output tile.  Q scaled by SCALE; V written transposed.

__global__ __launch_bounds__(256) void k_qkv(const unsigned short* __restrict__ xb,
                                             const unsigned short* __restrict__ WqT,
                                             const unsigned short* __restrict__ WkvT,
                                             unsigned short* __restrict__ Qb,
                                             unsigned short* __restrict__ Kb,
                                             unsigned short* __restrict__ VTb) {
  int wave = blockIdx.x * 8 + (threadIdx.x >> 5);
  int tr = wave / 96, tc = wave % 96;
  int row0 = tr << 4, col0 = tc << 4;

  const unsigned short* Bm;
  int bcol;
  if (col0 < 512) { Bm = WqT;  bcol = col0; }
  else            { Bm = WkvT; bcol = col0 - 512; }

  v8f acc = zero8();
#pragma unroll 4
  for (int k0 = 0; k0 < 512; k0 += 32) {
    v16bf a = load_a(xb, 512, row0, k0);
    v16bf b = load_b(Bm, 512, bcol, k0);
    acc = wmma_bf16(a, b, acc);
  }

  int lane = threadIdx.x & 31;
  int n = lane & 15, hi = lane >> 4;
  int bidx = row0 >> 11;        // batch
  int nrow = row0 & 2047;       // seq position base

  if (col0 < 512) {             // Q (pre-scaled)
    int h = col0 >> 6, d0 = col0 & 63;
    unsigned short* dst = Qb + ((size_t)(bidx * 8 + h) * 2048 + nrow) * 64 + d0 + n;
#pragma unroll
    for (int v = 0; v < 8; ++v) dst[(size_t)(v + 8 * hi) * 64] = f2bf(acc[v] * SCALE);
  } else if (col0 < 1024) {     // K stored [r][d]
    int c = col0 - 512; int h = c >> 6, d0 = c & 63;
    unsigned short* dst = Kb + ((size_t)(bidx * 8 + h) * 2048 + nrow) * 64 + d0 + n;
#pragma unroll
    for (int v = 0; v < 8; ++v) dst[(size_t)(v + 8 * hi) * 64] = f2bf(acc[v]);
  } else {                      // V stored transposed [d][r]
    int c = col0 - 1024; int h = c >> 6, d0 = c & 63;
    unsigned short* dst = VTb + ((size_t)(bidx * 8 + h) * 64 + d0 + n) * 2048 + nrow;
#pragma unroll
    for (int v = 0; v < 8; ++v) dst[v + 8 * hi] = f2bf(acc[v]);
  }
}

// ---------- fused flash attention with relative-position scores ----------
// One wave owns 16 query rows; loops keys in blocks of 32; online softmax.

__global__ __launch_bounds__(128) void k_attn(const unsigned short* __restrict__ Qb,
                                              const unsigned short* __restrict__ Kb,
                                              const unsigned short* __restrict__ VTb,
                                              const unsigned short* __restrict__ relb,
                                              unsigned short* __restrict__ AOb) {
  __shared__ float Elds[4][16][33];                        // per-wave E tile (fp32)
  __shared__ __align__(32) unsigned short Plds[4][16][32]; // per-wave P tile (bf16)

  int wid  = threadIdx.x >> 5;
  int wave = blockIdx.x * 4 + wid;          // 0..4095
  int bh   = wave >> 7;                     // b*8 + h
  int i0   = (wave & 127) << 4;             // query tile base

  const unsigned short* Q  = Qb  + (size_t)bh * 2048 * 64;
  const unsigned short* K  = Kb  + (size_t)bh * 2048 * 64;
  const unsigned short* VT = VTb + (size_t)bh * 64 * 2048;

  int lane = threadIdx.x & 31;
  int n = lane & 15, hi = lane >> 4;

  v16bf qa0 = load_a(Q, 64, i0, 0);         // Q tile, K-dim 0..31
  v16bf qa1 = load_a(Q, 64, i0, 32);        // Q tile, K-dim 32..63

  float rm[8], rs[8];
#pragma unroll
  for (int v = 0; v < 8; ++v) { rm[v] = -3.0e38f; rs[v] = 0.f; }
  v8f O0 = zero8(), O1 = zero8(), O2 = zero8(), O3 = zero8();

  for (int j0 = 0; j0 < 2048; j0 += 32) {
    // prefetch next key block (K: 32 rows x 128B; VT: 64 rows x 64B segments)
    if (j0 + 32 < 2048) {
      __builtin_prefetch(K + (size_t)(j0 + 32 + lane) * 64, 0, 1);
      __builtin_prefetch(VT + (size_t)lane * 2048 + j0 + 32, 0, 1);
      __builtin_prefetch(VT + (size_t)(lane + 32) * 2048 + j0 + 32, 0, 1);
    }

    v8f S0, S1;
#pragma unroll
    for (int t = 0; t < 2; ++t) {
      int jb = j0 + t * 16;
      // content scores: Q(16x64) @ K_tile^T
      v8f s = zero8();
      s = wmma_bf16(qa0, load_b(K, 64, jb, 0),  s);
      s = wmma_bf16(qa1, load_b(K, 64, jb, 32), s);
      // relative-position scores: E = Q @ relWindow^T, window base D0
      int D0 = i0 - jb + 497;               // dist index at window w=0
      v8f e0 = zero8(), e1 = zero8();
      e0 = wmma_bf16(qa0, load_rel(relb, D0, 0, 0),   e0);
      e0 = wmma_bf16(qa1, load_rel(relb, D0, 0, 32),  e0);
      e1 = wmma_bf16(qa0, load_rel(relb, D0, 16, 0),  e1);
      e1 = wmma_bf16(qa1, load_rel(relb, D0, 16, 32), e1);
#pragma unroll
      for (int v = 0; v < 8; ++v) {
        Elds[wid][v + 8 * hi][n]      = e0[v];
        Elds[wid][v + 8 * hi][16 + n] = e1[v];
      }
      // same-wave LDS is processed in-order: scatter-read pos[m,n] = E[m, m-n+15]
#pragma unroll
      for (int v = 0; v < 8; ++v) {
        int m = v + 8 * hi;
        s[v] += Elds[wid][m][m - n + 15];
      }
      if (t == 0) S0 = s; else S1 = s;
    }

    // online softmax over this 32-key block (row = (v,hi) group of 16 lanes)
#pragma unroll
    for (int v = 0; v < 8; ++v) {
      float bm = fmaxf(S0[v], S1[v]);
      bm = fmaxf(bm, __shfl_xor(bm, 1, 32));
      bm = fmaxf(bm, __shfl_xor(bm, 2, 32));
      bm = fmaxf(bm, __shfl_xor(bm, 4, 32));
      bm = fmaxf(bm, __shfl_xor(bm, 8, 32));
      float mnew  = fmaxf(rm[v], bm);
      float alpha = __expf(rm[v] - mnew);
      rm[v] = mnew;
      float p0 = __expf(S0[v] - mnew);
      float p1 = __expf(S1[v] - mnew);
      float ps = p0 + p1;
      ps += __shfl_xor(ps, 1, 32);
      ps += __shfl_xor(ps, 2, 32);
      ps += __shfl_xor(ps, 4, 32);
      ps += __shfl_xor(ps, 8, 32);
      rs[v] = rs[v] * alpha + ps;
      O0[v] *= alpha; O1[v] *= alpha; O2[v] *= alpha; O3[v] *= alpha;
      Plds[wid][v + 8 * hi][n]      = f2bf(p0);
      Plds[wid][v + 8 * hi][16 + n] = f2bf(p1);
    }

    // O += P(16x32) @ V(32x64): A-fragment of P via LDS transpose
    {
      const unsigned short* pp = &Plds[wid][lane & 15][(lane >> 4) << 4];
      ushort8 lo  = *(const ushort8*)pp;
      ushort8 hi8 = *(const ushort8*)(pp + 8);
      v16bf pa = make_frag(lo, hi8);
      O0 = wmma_bf16(pa, load_b(VT, 2048, 0,  j0), O0);
      O1 = wmma_bf16(pa, load_b(VT, 2048, 16, j0), O1);
      O2 = wmma_bf16(pa, load_b(VT, 2048, 32, j0), O2);
      O3 = wmma_bf16(pa, load_b(VT, 2048, 48, j0), O3);
    }
  }

  // finalize: divide by row sums, write bf16 attn-out [B, N, H*64]
  int b_ = bh >> 3, h = bh & 7;
  unsigned short* dst = AOb + (size_t)(b_ * 2048 + i0) * 512 + h * 64;
#pragma unroll
  for (int v = 0; v < 8; ++v) {
    int m = v + 8 * hi;
    float inv = 1.0f / rs[v];
    unsigned short* r = dst + (size_t)m * 512;
    r[n]      = f2bf(O0[v] * inv);
    r[16 + n] = f2bf(O1[v] * inv);
    r[32 + n] = f2bf(O2[v] * inv);
    r[48 + n] = f2bf(O3[v] * inv);
  }
}

// ---------- output projection: [8192,512] x [512,512] + bias ----------

__global__ __launch_bounds__(256) void k_out(const unsigned short* __restrict__ AOb,
                                             const unsigned short* __restrict__ WoT,
                                             const float* __restrict__ bo,
                                             float* __restrict__ out) {
  int wave = blockIdx.x * 8 + (threadIdx.x >> 5);
  int tr = wave >> 5, tc = wave & 31;
  int row0 = tr << 4, col0 = tc << 4;

  v8f acc = zero8();
#pragma unroll 4
  for (int k0 = 0; k0 < 512; k0 += 32) {
    acc = wmma_bf16(load_a(AOb, 512, row0, k0), load_b(WoT, 512, col0, k0), acc);
  }

  int lane = threadIdx.x & 31;
  int n = lane & 15, hi = lane >> 4;
  float bias = bo[col0 + n];
  float* dst = out + (size_t)row0 * 512 + col0 + n;
#pragma unroll
  for (int v = 0; v < 8; ++v) dst[(size_t)(v + 8 * hi) * 512] = acc[v] + bias;
}

// ---------- launcher ----------

extern "C" void kernel_launch(void* const* d_in, const int* in_sizes, int n_in,
                              void* d_out, int out_size, void* d_ws, size_t ws_size,
                              hipStream_t stream) {
  (void)in_sizes; (void)n_in; (void)out_size; (void)ws_size;
  const float* x   = (const float*)d_in[0];
  const float* Wq  = (const float*)d_in[1];
  const float* Wkv = (const float*)d_in[2];
  const float* Wo  = (const float*)d_in[3];
  const float* bo  = (const float*)d_in[4];
  const float* rel = (const float*)d_in[5];

  char* ws = (char*)d_ws;
  size_t off = 0;
  auto alloc = [&](size_t bytes) -> void* {
    void* p = ws + off;
    off = (off + bytes + 255) & ~(size_t)255;
    return p;
  };
  unsigned short* xb   = (unsigned short*)alloc((size_t)NROWS * DIM * 2);      // x bf16
  unsigned short* WqT  = (unsigned short*)alloc((size_t)512 * 512 * 2);        // Wq^T
  unsigned short* WkvT = (unsigned short*)alloc((size_t)1024 * 512 * 2);       // Wkv^T
  unsigned short* WoT  = (unsigned short*)alloc((size_t)512 * 512 * 2);        // Wo^T
  unsigned short* relb = (unsigned short*)alloc((size_t)NREL * 64 * 2);        // rel bf16
  unsigned short* Qb   = (unsigned short*)alloc((size_t)NB * HEADS * NSEQ * 64 * 2);
  unsigned short* Kb   = (unsigned short*)alloc((size_t)NB * HEADS * NSEQ * 64 * 2);
  unsigned short* VTb  = (unsigned short*)alloc((size_t)NB * HEADS * 64 * NSEQ * 2);
  unsigned short* AOb  = (unsigned short*)alloc((size_t)NROWS * INNER * 2);

  int nx = NROWS * DIM;
  k_f2bf <<<(nx + 255) / 256, 256, 0, stream>>>(x, xb, nx);
  k_f2bfT<<<(512 * 512  + 255) / 256, 256, 0, stream>>>(Wq,  WqT,  512, 512);
  k_f2bfT<<<(512 * 1024 + 255) / 256, 256, 0, stream>>>(Wkv, WkvT, 512, 1024);
  k_f2bfT<<<(512 * 512  + 255) / 256, 256, 0, stream>>>(Wo,  WoT,  512, 512);
  k_f2bf <<<(NREL * 64 + 255) / 256, 256, 0, stream>>>(rel, relb, NREL * 64);

  k_qkv <<<6144, 256, 0, stream>>>(xb, WqT, WkvT, Qb, Kb, VTb);
  k_attn<<<1024, 128, 0, stream>>>(Qb, Kb, VTb, relb, AOb);
  k_out <<<2048, 256, 0, stream>>>(AOb, WoT, bo, (float*)d_out);
}
